// ConvNetSelfAttn_58213986730114
// MI455X (gfx1250) — matrix operand order repaired and checked by reading, not compiled, not run on hardware
//
#include <hip/hip_runtime.h>
#include <hip/hip_bf16.h>
#include <math.h>

// ---------------------------------------------------------------------------
// ConvNet + self-attention for MI455X (gfx1250, wave32, WMMA).
//
// Layouts:
//   h buffers : (B=8, 304, N=2048) f32, rows >= Cout unused.
//   qt, kt    : (B, N, 64) f16, channel dim zero-padded C8 -> 64.
//   vh        : (B, 304, N) f16, rows >= Cout zeroed.
// Attention kernel: one workgroup per (batch, 16-row tile). Full 16x2048
// energy stripe lives in LDS (f32) + f16 probabilities (193 KB total,
// exploiting CDNA5's 320 KB/WGP LDS). Both GEMMs use
// v_wmma_f32_16x16x32_f16; softmax normalization folded into epilogue.
// Stage 3 uses two independent WMMA accumulators to double in-wave ILP
// (LDS-limited to 1 WG/WGP -> 2 waves/SIMD, so chain latency must be
// hidden inside the wave).
// ---------------------------------------------------------------------------

typedef __attribute__((ext_vector_type(16))) _Float16 v16h;
typedef __attribute__((ext_vector_type(8)))  float    v8f;

#define NPTS   2048
#define BATCH  8
#define CMAX   304          // padded channel stride for h / vh buffers

// ---------------------------------------------------------------------------
// conv1x1 + ReLU:  out[b,c,n] = relu( sum_cc W[c,cc]*in(b,cc,n) + bias[c] )
// Input addressed with generic strides so block 0 can read x in (B,N,C).
// grid: (2048/256, Cout, B), block 256.
// ---------------------------------------------------------------------------
__global__ void conv_relu_kernel(const float* __restrict__ in,
                                 const float* __restrict__ w,
                                 const float* __restrict__ bias,
                                 float* __restrict__ out,
                                 int Cin, long in_sb, long in_sc, long in_sn) {
    int n = blockIdx.x * 256 + threadIdx.x;
    int c = blockIdx.y;
    int b = blockIdx.z;
    const float* wr = w + (size_t)c * Cin;
    const float* ip = in + (long)b * in_sb + (long)n * in_sn;
    float acc = bias[c];
    for (int cc = 0; cc < Cin; ++cc)
        acc += wr[cc] * ip[(long)cc * in_sc];
    out[((size_t)(b * CMAX + c)) * NPTS + n] = fmaxf(acc, 0.0f);
}

// ---------------------------------------------------------------------------
// q/k projections, written transposed + padded as f16:
//   qt[b,n,c8] = sum_cc qw[c8,cc]*h[b,cc,n] + qb[c8]   (c8 in [C8,64) -> 0)
// block 256 = 4 tokens x 64 channel slots; grid (2048/4, B).
// ---------------------------------------------------------------------------
__global__ void qk_kernel(const float* __restrict__ h,
                          const float* __restrict__ qw, const float* __restrict__ qb,
                          const float* __restrict__ kw, const float* __restrict__ kb,
                          _Float16* __restrict__ qt, _Float16* __restrict__ kt,
                          int Cout, int C8) {
    int c8 = threadIdx.x & 63;
    int n  = blockIdx.x * 4 + (threadIdx.x >> 6);
    int b  = blockIdx.y;
    float accq = 0.0f, acck = 0.0f;
    if (c8 < C8) {
        accq = qb[c8];
        acck = kb[c8];
        const float* qwr = qw + (size_t)c8 * Cout;
        const float* kwr = kw + (size_t)c8 * Cout;
        const float* hp  = h + (size_t)b * CMAX * NPTS + n;
        for (int cc = 0; cc < Cout; ++cc) {
            float hv = hp[(size_t)cc * NPTS];
            accq += qwr[cc] * hv;
            acck += kwr[cc] * hv;
        }
    }
    size_t o = ((size_t)b * NPTS + n) * 64 + c8;
    qt[o] = (_Float16)accq;
    kt[o] = (_Float16)acck;
}

// ---------------------------------------------------------------------------
// v projection to f16 (natural (C,N) layout == B^T layout for 2nd GEMM):
// grid (2048/256, 304, B), block 256. Rows >= Cout zeroed.
// ---------------------------------------------------------------------------
__global__ void v_kernel(const float* __restrict__ h,
                         const float* __restrict__ vw, const float* __restrict__ vb,
                         _Float16* __restrict__ vh, int Cout) {
    int n = blockIdx.x * 256 + threadIdx.x;
    int c = blockIdx.y;
    int b = blockIdx.z;
    float acc = 0.0f;
    if (c < Cout) {
        acc = vb[c];
        const float* wr = vw + (size_t)c * Cout;
        const float* hp = h + (size_t)b * CMAX * NPTS + n;
        for (int cc = 0; cc < Cout; ++cc)
            acc += wr[cc] * hp[(size_t)cc * NPTS];
    }
    vh[((size_t)(b * CMAX + c)) * NPTS + n] = (_Float16)acc;
}

// ---------------------------------------------------------------------------
// Attention: one WG (256 thr = 8 waves) per (b, 16-row tile n0).
//  Stage 1: energy E[16,2048] via WMMA (K = 64 padded head dim) -> LDS f32.
//  Stage 2: exact row softmax in LDS; unnormalized probs -> LDS f16.
//  Stage 3: att_out[n, c] = sum_m P[n,m] * v[c,m] via WMMA, K = 2048,
//           dual accumulators for ILP.
//  Epilogue: hOut = gamma * att/rowsum + hIn.
// Dynamic LDS: 131072 (E) + 65536 (P) + 1024 (red) + 128 (rowstat) bytes.
// ---------------------------------------------------------------------------
__global__ __launch_bounds__(256)
void attn_kernel(const float* __restrict__ hIn,
                 const _Float16* __restrict__ qt,
                 const _Float16* __restrict__ kt,
                 const _Float16* __restrict__ vh,
                 const float* __restrict__ gamma,
                 float* __restrict__ hOut, int Cout) {
    extern __shared__ char smem[];
    float*    E       = (float*)smem;                       // 16 x 2048 f32
    _Float16* P       = (_Float16*)(smem + 131072);         // 16 x 2048 f16
    float*    red     = (float*)(smem + 131072 + 65536);    // 16 x 16
    float*    rowstat = red + 256;                          // 16 max + 16 invsum

    const int tid  = threadIdx.x;
    const int wave = tid >> 5;
    const int lane = tid & 31;
    const int half = lane >> 4;
    const int l16  = lane & 15;
    const int n0   = blockIdx.x * 16;
    const int b    = blockIdx.y;

    // ---- Stage 1: energy ----
    // A fragments (rows n0..n0+15 of q^T, K = 0..63), kept for all 128 m-tiles
    v16h a0, a1;
    {
        const _Float16* qrow = qt + ((size_t)(b * NPTS + n0 + l16)) * 64;
#pragma unroll
        for (int e = 0; e < 16; ++e) {
            int j = e >> 1, p = e & 1;
            int k = 16 * half + e; (void)j; (void)p;
            a0[e] = qrow[k];
            a1[e] = qrow[32 + k];
        }
    }
    for (int t = wave; t < 128; t += 8) {
        int m0 = t * 16;
        const _Float16* krow =
            kt + ((size_t)(b * NPTS + m0 + l16)) * 64 + 16 * half;
        v16h b0, b1;
#pragma unroll
        for (int e = 0; e < 16; ++e) { b0[e] = krow[e]; b1[e] = krow[32 + e]; }
        v8f c = {};
        c = __builtin_amdgcn_wmma_f32_16x16x32_f16(false, a0, false, b0,
                                                   (short)0, c, false, false);
        c = __builtin_amdgcn_wmma_f32_16x16x32_f16(false, a1, false, b1,
                                                   (short)0, c, false, false);
#pragma unroll
        for (int r = 0; r < 8; ++r)
            E[(size_t)(r + 8 * half) * NPTS + m0 + l16] = c[r];
    }
    __syncthreads();

    // ---- Stage 2: softmax (exact, two-pass; normalization deferred) ----
    {
        int row = tid >> 4, sub = tid & 15;
        const float* Er = E + (size_t)row * NPTS;
        float mx = -3.4e38f;
        for (int m = sub; m < NPTS; m += 16) mx = fmaxf(mx, Er[m]);
        red[row * 16 + sub] = mx;
        __syncthreads();
        if (sub == 0) {
            float m2 = red[row * 16];
            for (int i = 1; i < 16; ++i) m2 = fmaxf(m2, red[row * 16 + i]);
            rowstat[row] = m2;
        }
        __syncthreads();
        float rmax = rowstat[row];
        float s = 0.0f;
        _Float16* Pr = P + (size_t)row * NPTS;
        for (int m = sub; m < NPTS; m += 16) {
            float ev = __expf(Er[m] - rmax);
            Pr[m] = (_Float16)ev;
            s += ev;
        }
        __syncthreads();
        red[row * 16 + sub] = s;
        __syncthreads();
        if (sub == 0) {
            float t2 = 0.0f;
            for (int i = 0; i < 16; ++i) t2 += red[row * 16 + i];
            rowstat[16 + row] = 1.0f / t2;
        }
        __syncthreads();
    }

    // ---- Stage 3: att_out = P x v^T, K = 2048, dual-accumulator WMMA ----
    const float g = gamma[0];
    const int Cpad = (Cout + 15) & ~15;
    const int nct = Cpad >> 4;
    const _Float16* Pr = P + (size_t)l16 * NPTS;   // A row = l16
    for (int ct = wave; ct < nct; ct += 8) {
        int c0 = ct * 16;
        const _Float16* vrow =
            vh + ((size_t)(b * CMAX + c0 + l16)) * NPTS + 16 * half;
        v8f accA = {}, accB = {};
        for (int ks = 0; ks < 64; ks += 2) {
            v16h aE, bE, aO, bO;
#pragma unroll
            for (int e = 0; e < 16; ++e) {
                int j = e >> 1, p = e & 1;
                int k = 16 * half + e; (void)j; (void)p;
                aE[e] = Pr[ks * 32 + k];
                aO[e] = Pr[ks * 32 + 32 + k];
                bE[e] = vrow[ks * 32 + e];        // 16 contiguous f16 / lane
                bO[e] = vrow[ks * 32 + 32 + e];
            }
            accA = __builtin_amdgcn_wmma_f32_16x16x32_f16(false, aE, false, bE,
                                                          (short)0, accA,
                                                          false, false);
            accB = __builtin_amdgcn_wmma_f32_16x16x32_f16(false, aO, false, bO,
                                                          (short)0, accB,
                                                          false, false);
        }
        v8f acc = accA + accB;
        int c = c0 + l16;
        if (c < Cout) {
            size_t base = ((size_t)(b * CMAX + c)) * NPTS + n0;
#pragma unroll
            for (int r = 0; r < 8; ++r) {
                int nl = r + 8 * half;
                hOut[base + nl] =
                    g * acc[r] * rowstat[16 + nl] + hIn[base + nl];
            }
        }
    }
}

// ---------------------------------------------------------------------------
// Max-pool over N, then 2-way linear head. grid B, block 256.
// ---------------------------------------------------------------------------
__global__ void pool_fc_kernel(const float* __restrict__ h,
                               const float* __restrict__ w,
                               const float* __restrict__ bias,
                               float* __restrict__ out, int Cout) {
    __shared__ float pooled[CMAX];
    int b = blockIdx.x, tid = threadIdx.x;
    for (int c = tid; c < Cout; c += 256) {
        const float* hr = h + ((size_t)(b * CMAX + c)) * NPTS;
        float mx = hr[0];
        for (int n = 1; n < NPTS; ++n) mx = fmaxf(mx, hr[n]);
        pooled[c] = mx;
    }
    __syncthreads();
    if (tid < 2) {
        float acc = bias[tid];
        const float* wr = w + (size_t)tid * Cout;
        for (int c = 0; c < Cout; ++c) acc += pooled[c] * wr[c];
        out[b * 2 + tid] = acc;
    }
}

// ---------------------------------------------------------------------------
// Orchestration. Input order (setup_inputs dict order, leaves in insertion
// order): x, then 6 blocks x {cw,cb,qw,qb,kw,kb,vw,vb,gamma}, then out_w,
// out_b.
// ---------------------------------------------------------------------------
extern "C" void kernel_launch(void* const* d_in, const int* in_sizes, int n_in,
                              void* d_out, int out_size, void* d_ws,
                              size_t ws_size, hipStream_t stream) {
    (void)in_sizes; (void)n_in; (void)out_size; (void)ws_size;
    static const int CH[7] = {512, 100, 100, 200, 200, 300, 300};

    const float* x = (const float*)d_in[0];

    char* ws = (char*)d_ws;
    size_t off = 0;
    const size_t hbytes = (size_t)BATCH * CMAX * NPTS * sizeof(float);
    float* hA = (float*)(ws + off); off += hbytes;   // conv output
    float* hB = (float*)(ws + off); off += hbytes;   // attention output
    _Float16* qt = (_Float16*)(ws + off); off += (size_t)BATCH * NPTS * 64 * 2;
    _Float16* kt = (_Float16*)(ws + off); off += (size_t)BATCH * NPTS * 64 * 2;
    _Float16* vh = (_Float16*)(ws + off); off += (size_t)BATCH * CMAX * NPTS * 2;

    const size_t attn_smem = 131072 + 65536 + 1024 + 128;  // ~193 KB < 320 KB

    for (int i = 0; i < 6; ++i) {
        const int Cin = CH[i], Cout = CH[i + 1], C8 = Cout / 8;
        const float* cw    = (const float*)d_in[1 + 9 * i + 0];
        const float* cb    = (const float*)d_in[1 + 9 * i + 1];
        const float* qw    = (const float*)d_in[1 + 9 * i + 2];
        const float* qb    = (const float*)d_in[1 + 9 * i + 3];
        const float* kw    = (const float*)d_in[1 + 9 * i + 4];
        const float* kb    = (const float*)d_in[1 + 9 * i + 5];
        const float* vw    = (const float*)d_in[1 + 9 * i + 6];
        const float* vb    = (const float*)d_in[1 + 9 * i + 7];
        const float* gamma = (const float*)d_in[1 + 9 * i + 8];

        // conv1x1 + relu -> hA
        const float* inP;
        long sb, sc, sn;
        if (i == 0) { inP = x;  sb = (long)NPTS * 512; sc = 1;    sn = 512; }
        else        { inP = hB; sb = (long)CMAX * NPTS; sc = NPTS; sn = 1;  }
        conv_relu_kernel<<<dim3(NPTS / 256, Cout, BATCH), 256, 0, stream>>>(
            inP, cw, cb, hA, Cin, sb, sc, sn);

        // q/k transposed+padded f16, v f16
        qk_kernel<<<dim3(NPTS / 4, BATCH), 256, 0, stream>>>(
            hA, qw, qb, kw, kb, qt, kt, Cout, C8);
        v_kernel<<<dim3(NPTS / 256, CMAX, BATCH), 256, 0, stream>>>(
            hA, vw, vb, vh, Cout);

        // attention + residual -> hB
        attn_kernel<<<dim3(NPTS / 16, BATCH), 256, attn_smem, stream>>>(
            hA, qt, kt, vh, gamma, hB, Cout);
    }

    const float* out_w = (const float*)d_in[55];
    const float* out_b = (const float*)d_in[56];
    pool_fc_kernel<<<BATCH, 256, 0, stream>>>(hB, out_w, out_b,
                                              (float*)d_out, CH[6]);
}
